// PruneModel_20804821581907
// MI455X (gfx1250) — matrix operand order repaired and checked by reading, not compiled, not run on hardware
//
#include <hip/hip_runtime.h>
#include <stdint.h>

// ---------------------------------------------------------------------------
// Types for CDNA5 WMMA
// ---------------------------------------------------------------------------
typedef __attribute__((ext_vector_type(16))) __bf16 bf16x16;
typedef __attribute__((ext_vector_type(8)))  float  floatx8;

#define DCH 128      // hidden dim
#define GEN_EPS 1e-7f
#define BN_EPS  1e-5f

// ---------------------------------------------------------------------------
// float atomic max via signed/unsigned integer ordering trick (init = -inf)
// ---------------------------------------------------------------------------
__device__ __forceinline__ void atomicMaxF(float* a, float v) {
  if (v >= 0.f) atomicMax((int*)a, __float_as_int(v));
  else          atomicMin((unsigned int*)a, __float_as_uint(v));
}

// ---------------------------------------------------------------------------
// Elementwise helpers
// ---------------------------------------------------------------------------
__global__ void k_fill(float* p, float v, long long n) {
  long long st = (long long)gridDim.x * blockDim.x;
  for (long long i = (long long)blockIdx.x * blockDim.x + threadIdx.x; i < n; i += st)
    p[i] = v;
}

__global__ void k_init3(float* mx, float* den, float* agg, long long n) {
  long long st = (long long)gridDim.x * blockDim.x;
  for (long long i = (long long)blockIdx.x * blockDim.x + threadIdx.x; i < n; i += st) {
    mx[i]  = __int_as_float(0xff800000);  // -inf
    den[i] = 0.f;
    agg[i] = 0.f;
  }
}

__global__ void k_maskmul_f32(const float* __restrict__ a, const float* __restrict__ m,
                              float* __restrict__ o, long long n) {
  long long st = (long long)gridDim.x * blockDim.x;
  for (long long i = (long long)blockIdx.x * blockDim.x + threadIdx.x; i < n; i += st)
    o[i] = m ? a[i] * m[i] : a[i];
}

__global__ void k_maskmul_bf16(const float* __restrict__ a, const float* __restrict__ m,
                               __bf16* __restrict__ o, long long n) {
  long long st = (long long)gridDim.x * blockDim.x;
  for (long long i = (long long)blockIdx.x * blockDim.x + threadIdx.x; i < n; i += st)
    o[i] = (__bf16)(m ? a[i] * m[i] : a[i]);
}

__global__ void k_add_bf16(const float* __restrict__ a, const float* __restrict__ b,
                           __bf16* __restrict__ o, long long n) {
  long long st = (long long)gridDim.x * blockDim.x;
  for (long long i = (long long)blockIdx.x * blockDim.x + threadIdx.x; i < n; i += st)
    o[i] = (__bf16)(a[i] + b[i]);
}

__global__ void k_cvt_bf16(const float* __restrict__ a, __bf16* __restrict__ o, long long n) {
  long long st = (long long)gridDim.x * blockDim.x;
  for (long long i = (long long)blockIdx.x * blockDim.x + threadIdx.x; i < n; i += st)
    o[i] = (__bf16)a[i];
}

// ---------------------------------------------------------------------------
// Fused small-K encoder: y[r,c] = dot(in[r,:K], Wm[c,:K]) + bm[c]
//   stats variant accumulates per-channel sum / sumsq via LDS (ds_add_f32)
//   apply variant applies precomputed BN scale/shift and writes f32 and/or bf16
// ---------------------------------------------------------------------------
__global__ __launch_bounds__(256) void k_encode_stats(
    const float* __restrict__ in, const float* __restrict__ Wm,
    const float* __restrict__ bm, long long total, int K,
    float* __restrict__ stats /* [128] sum, [128] sumsq */) {
  __shared__ float ssum[DCH], ssq[DCH];
  if (threadIdx.x < DCH) { ssum[threadIdx.x] = 0.f; ssq[threadIdx.x] = 0.f; }
  __syncthreads();
  long long st = (long long)gridDim.x * blockDim.x;
  for (long long i = (long long)blockIdx.x * blockDim.x + threadIdx.x; i < total; i += st) {
    int c = (int)(i & (DCH - 1));
    long long r = i >> 7;
    const float* xr = in + r * K;
    const float* wr = Wm + (long long)c * K;
    float acc = bm[c];
    for (int k = 0; k < K; ++k) acc += xr[k] * wr[k];
    atomicAdd(&ssum[c], acc);
    atomicAdd(&ssq[c], acc * acc);
  }
  __syncthreads();
  if (threadIdx.x < DCH) {
    atomicAdd(&stats[threadIdx.x], ssum[threadIdx.x]);
    atomicAdd(&stats[DCH + threadIdx.x], ssq[threadIdx.x]);
  }
}

__global__ __launch_bounds__(256) void k_encode_apply(
    const float* __restrict__ in, const float* __restrict__ Wm,
    const float* __restrict__ bm, const float* __restrict__ ss,
    float* __restrict__ outf, __bf16* __restrict__ outb,
    long long total, int K) {
  long long st = (long long)gridDim.x * blockDim.x;
  for (long long i = (long long)blockIdx.x * blockDim.x + threadIdx.x; i < total; i += st) {
    int c = (int)(i & (DCH - 1));
    long long r = i >> 7;
    const float* xr = in + r * K;
    const float* wr = Wm + (long long)c * K;
    float acc = bm[c];
    for (int k = 0; k < K; ++k) acc += xr[k] * wr[k];
    float y = acc * ss[c] + ss[DCH + c];
    if (outf) outf[i] = y;
    if (outb) outb[i] = (__bf16)y;
  }
}

// ---------------------------------------------------------------------------
// BN finalize: scale = g * rsqrt(var + eps), shift = b - mean*scale
// ---------------------------------------------------------------------------
__global__ void k_bn_finalize(const float* __restrict__ stats, const float* __restrict__ g,
                              const float* __restrict__ b, float invn,
                              float* __restrict__ ss, int C) {
  int c = blockIdx.x * blockDim.x + threadIdx.x;
  if (c >= C) return;
  float mean = stats[c] * invn;
  float var  = stats[C + c] * invn - mean * mean;
  float sc   = g[c] * rsqrtf(var + BN_EPS);
  ss[c]     = sc;
  ss[C + c] = b[c] - mean * sc;
}

// ---------------------------------------------------------------------------
// Column stats over [rows, C] with LDS two-stage reduction (C = 128 or 256)
// ---------------------------------------------------------------------------
__global__ __launch_bounds__(256) void k_colstats(
    const float* __restrict__ x, long long total, int C, float* __restrict__ stats) {
  extern __shared__ float sm[];  // 2*C floats
  for (int j = threadIdx.x; j < 2 * C; j += blockDim.x) sm[j] = 0.f;
  __syncthreads();
  long long st = (long long)gridDim.x * blockDim.x;
  for (long long i = (long long)blockIdx.x * blockDim.x + threadIdx.x; i < total; i += st) {
    float v = x[i];
    int c = (int)(i & (C - 1));
    atomicAdd(&sm[c], v);
    atomicAdd(&sm[C + c], v * v);
  }
  __syncthreads();
  for (int j = threadIdx.x; j < 2 * C; j += blockDim.x) atomicAdd(&stats[j], sm[j]);
}

__global__ void k_bn_relu_bf16(const float* __restrict__ x, const float* __restrict__ ss,
                               __bf16* __restrict__ o, long long n, int C) {
  long long st = (long long)gridDim.x * blockDim.x;
  for (long long i = (long long)blockIdx.x * blockDim.x + threadIdx.x; i < n; i += st) {
    int c = (int)(i & (C - 1));
    o[i] = (__bf16)fmaxf(x[i] * ss[c] + ss[C + c], 0.f);
  }
}

__global__ void k_bn_relu_add(const float* __restrict__ x, const float* __restrict__ ss,
                              float* __restrict__ h, long long n) {
  long long st = (long long)gridDim.x * blockDim.x;
  for (long long i = (long long)blockIdx.x * blockDim.x + threadIdx.x; i < n; i += st) {
    int c = (int)(i & (DCH - 1));
    h[i] += fmaxf(x[i] * ss[c] + ss[DCH + c], 0.f);
  }
}

// ---------------------------------------------------------------------------
// Edge passes: msg = relu(h[src]+e)+eps ; softmax aggregation to dst
// Each thread handles 4 channels of one edge (E*32 threads per pass).
// ---------------------------------------------------------------------------
__global__ __launch_bounds__(256) void k_edge_max(
    const __bf16* __restrict__ e, const float* __restrict__ h,
    const int* __restrict__ src, const int* __restrict__ dst,
    const float* __restrict__ tptr, int layer, float* __restrict__ mx,
    long long total) {
  long long i = (long long)blockIdx.x * blockDim.x + threadIdx.x;
  if (i >= total) return;
  long long ed = i >> 5;
  int c0 = ((int)i & 31) << 2;
  int s = src[ed], d = dst[ed];
  float t = tptr[layer];
  const float*  hp = h + (long long)s * DCH + c0;
  const __bf16* ep = e + ed * DCH + c0;
  __builtin_prefetch(ep + DCH * 8, 0, 1);  // global_prefetch for edges ahead
#pragma unroll
  for (int j = 0; j < 4; ++j) {
    float m = fmaxf(hp[j] + (float)ep[j], 0.f) + GEN_EPS;
    atomicMaxF(&mx[(long long)d * DCH + c0 + j], m * t);
  }
}

__global__ __launch_bounds__(256) void k_edge_den(
    const __bf16* __restrict__ e, const float* __restrict__ h,
    const int* __restrict__ src, const int* __restrict__ dst,
    const float* __restrict__ tptr, int layer,
    const float* __restrict__ mx, float* __restrict__ den, long long total) {
  long long i = (long long)blockIdx.x * blockDim.x + threadIdx.x;
  if (i >= total) return;
  long long ed = i >> 5;
  int c0 = ((int)i & 31) << 2;
  int s = src[ed], d = dst[ed];
  float t = tptr[layer];
  const float*  hp = h + (long long)s * DCH + c0;
  const __bf16* ep = e + ed * DCH + c0;
#pragma unroll
  for (int j = 0; j < 4; ++j) {
    float m  = fmaxf(hp[j] + (float)ep[j], 0.f) + GEN_EPS;
    float ex = __expf(m * t - mx[(long long)d * DCH + c0 + j]);
    atomicAdd(&den[(long long)d * DCH + c0 + j], ex);
  }
}

__global__ __launch_bounds__(256) void k_edge_agg(
    const __bf16* __restrict__ e, const float* __restrict__ h,
    const int* __restrict__ src, const int* __restrict__ dst,
    const float* __restrict__ tptr, int layer,
    const float* __restrict__ mx, const float* __restrict__ den,
    float* __restrict__ agg, long long total) {
  long long i = (long long)blockIdx.x * blockDim.x + threadIdx.x;
  if (i >= total) return;
  long long ed = i >> 5;
  int c0 = ((int)i & 31) << 2;
  int s = src[ed], d = dst[ed];
  float t = tptr[layer];
  const float*  hp = h + (long long)s * DCH + c0;
  const __bf16* ep = e + ed * DCH + c0;
#pragma unroll
  for (int j = 0; j < 4; ++j) {
    long long o = (long long)d * DCH + c0 + j;
    float m  = fmaxf(hp[j] + (float)ep[j], 0.f) + GEN_EPS;
    float ex = __expf(m * t - mx[o]);
    float w  = ex / (den[o] + 1e-16f);
    atomicAdd(&agg[o], m * w);
  }
}

// ---------------------------------------------------------------------------
// WMMA GEMM: C[M,Nout] = A[M,K](bf16, row-major) @ W[Nout,K](bf16, row-major)^T
// Block = 256 threads = 8 waves in a 4x2 grid of 16x16 tiles -> 64x32 per block.
// Uses v_wmma_f32_16x16x32_bf16 with the documented CDNA5 VGPR layouts.
// ---------------------------------------------------------------------------
__global__ __launch_bounds__(256) void k_gemm_bf16(
    const __bf16* __restrict__ A, const __bf16* __restrict__ W,
    const float* __restrict__ bias, float* __restrict__ C,
    int M, int K, int Nout) {
  const int lane = threadIdx.x & 31;
  const int wave = threadIdx.x >> 5;
  const int wr = wave >> 1, wc = wave & 1;
  const int tile_m = blockIdx.x * 64 + wr * 16;
  const int tile_n = blockIdx.y * 32 + wc * 16;
  if (tile_m >= M) return;
  const int row   = min(tile_m + (lane & 15), M - 1);
  const int col   = tile_n + (lane & 15);
  const int ahalf = (lane >> 4) << 3;   // A K-pair sub-offset: 0 or 8
  const int bhalf = (lane >> 4) << 3;   // B K-pair sub-offset: 0 or 8
  floatx8 acc = {};
  for (int k0 = 0; k0 < K; k0 += 32) {
    union { bf16x16 v; uint32_t u[8]; } a, b;
    const uint32_t* Ap = (const uint32_t*)(A + (size_t)row * K + k0);
#pragma unroll
    for (int j = 0; j < 8; ++j) {
      // A 16x32 bf16 layout: VGPR j holds K pair {kp,kp+1}
      int kp = ((j & 4) << 2) + ((j & 3) << 1) + ahalf;
      a.u[j] = Ap[kp >> 1];
    }
    const uint32_t* Bp = (const uint32_t*)(W + (size_t)col * K + k0);
#pragma unroll
    for (int j = 0; j < 8; ++j) b.u[j] = Bp[(((j & 4) << 2) + ((j & 3) << 1) + bhalf) >> 1];  // B 32x16: same K pairs as A
    acc = __builtin_amdgcn_wmma_f32_16x16x32_bf16(false, a.v, false, b.v,
                                                  (short)0, acc, false, false);
  }
  const float bv   = bias ? bias[col] : 0.f;
  const int  rbase = tile_m + ((lane >> 4) << 3);
#pragma unroll
  for (int v = 0; v < 8; ++v) {
    int r = rbase + v;
    if (r < M) C[(size_t)r * Nout + col] = acc[v] + bv;
  }
}

// ---------------------------------------------------------------------------
// Pooling + readout
// ---------------------------------------------------------------------------
__global__ void k_pool(const float* __restrict__ h, const int* __restrict__ batch,
                       float* __restrict__ pooled, float* __restrict__ cnt, long long total) {
  long long st = (long long)gridDim.x * blockDim.x;
  for (long long i = (long long)blockIdx.x * blockDim.x + threadIdx.x; i < total; i += st) {
    int c = (int)(i & (DCH - 1));
    long long nidx = i >> 7;
    int b = batch[nidx];
    atomicAdd(&pooled[(long long)b * DCH + c], h[i]);
    if (c == 0) atomicAdd(&cnt[b], 1.f);
  }
}

__global__ __launch_bounds__(256) void k_final(
    const float* __restrict__ pooled, const float* __restrict__ cnt,
    const float* __restrict__ foWm, const float* __restrict__ fob,
    const float* __restrict__ fomb, float* __restrict__ out, int G) {
  int lane = threadIdx.x & 31;
  int g = blockIdx.x * 8 + (threadIdx.x >> 5);  // one wave32 per graph
  if (g >= G) return;
  float inv = 1.f / fmaxf(cnt[g], 1.f);
  float s = 0.f;
#pragma unroll
  for (int j = 0; j < 4; ++j) {
    int c = lane * 4 + j;
    s += pooled[(long long)g * DCH + c] * inv * foWm[c];
  }
  for (int off = 16; off; off >>= 1) s += __shfl_down(s, off, 32);
  if (lane == 0) {
    float z = s + fob[0] * fomb[0];
    out[g] = 1.f / (1.f + __expf(-z));
  }
}

// ---------------------------------------------------------------------------
// Host driver
// ---------------------------------------------------------------------------
static inline int nblk(long long n) {
  long long b = (n + 255) / 256;
  return (int)(b < (1 << 20) ? b : (1 << 20));
}

extern "C" void kernel_launch(void* const* d_in, const int* in_sizes, int n_in,
                              void* d_out, int out_size, void* d_ws, size_t ws_size,
                              hipStream_t stream) {
  const int N = in_sizes[0] / 16;       // 40000
  const int E = in_sizes[2] / 8;        // 640000
  const int G = out_size;               // 256
  const int L = in_sizes[16];           // 4
  const long long NH = (long long)N * DCH;
  const long long EH = (long long)E * DCH;

  // ---- inputs ----
  const float* x         = (const float*)d_in[0];
  const int*   eidx      = (const int*)d_in[1];
  const int*   src       = eidx;
  const int*   dst       = eidx + E;
  const float* eattr     = (const float*)d_in[2];
  const int*   batch     = (const int*)d_in[3];
  const float* enc_n_W   = (const float*)d_in[4];
  const float* enc_n_b   = (const float*)d_in[5];
  const float* enc_n_mW  = (const float*)d_in[6];
  const float* enc_n_mb  = (const float*)d_in[7];
  const float* enc_e_W   = (const float*)d_in[8];
  const float* enc_e_b   = (const float*)d_in[9];
  const float* enc_e_mW  = (const float*)d_in[10];
  const float* enc_e_mb  = (const float*)d_in[11];
  const float* bn_n_g    = (const float*)d_in[12];
  const float* bn_n_b    = (const float*)d_in[13];
  const float* bn_e_g    = (const float*)d_in[14];
  const float* bn_e_b    = (const float*)d_in[15];
  const float* tvec      = (const float*)d_in[16];
  const float* cW1       = (const float*)d_in[17];
  const float* cb1       = (const float*)d_in[18];
  const float* cg1       = (const float*)d_in[19];
  const float* cbt1      = (const float*)d_in[20];
  const float* cW2       = (const float*)d_in[21];
  const float* cb2       = (const float*)d_in[22];
  const float* mW1       = (const float*)d_in[23];
  const float* mb1       = (const float*)d_in[24];
  const float* mM1w      = (const float*)d_in[25];
  const float* mM1b      = (const float*)d_in[26];
  const float* mg1       = (const float*)d_in[27];
  const float* mbt1      = (const float*)d_in[28];
  const float* mW2       = (const float*)d_in[29];
  const float* mb2       = (const float*)d_in[30];
  const float* mM2w      = (const float*)d_in[31];
  const float* mM2b      = (const float*)d_in[32];
  const float* mg2       = (const float*)d_in[33];
  const float* mbt2      = (const float*)d_in[34];
  const float* fo_W      = (const float*)d_in[35];
  const float* fo_b      = (const float*)d_in[36];
  const float* fo_mW     = (const float*)d_in[37];
  const float* fo_mb     = (const float*)d_in[38];
  float* out             = (float*)d_out;

  // ---- workspace layout (256B-aligned bump allocator) ----
  char* wsb = (char*)d_ws;
  size_t off = 0;
  auto alloc = [&](size_t bytes) -> void* {
    off = (off + 255) & ~(size_t)255;
    void* p = wsb + off;
    off += bytes;
    return p;
  };
  __bf16* e_bf   = (__bf16*)alloc(EH * 2);
  float*  h      = (float*)alloc(NH * 4);
  float*  mx     = (float*)alloc(NH * 4);
  float*  den    = (float*)alloc(NH * 4);
  float*  agg    = (float*)alloc(NH * 4);
  __bf16* u_bf   = (__bf16*)alloc(NH * 2);
  float*  t1     = (float*)alloc((long long)N * 256 * 4);
  __bf16* t1_bf  = (__bf16*)alloc((long long)N * 256 * 2);
  float*  t2     = (float*)alloc(NH * 4);
  __bf16* t2_bf  = (__bf16*)alloc(NH * 2);
  float*  t4     = (float*)alloc(NH * 4);
  float*  stats  = (float*)alloc(512 * 4);
  float*  ss     = (float*)alloc(512 * 4);
  float*  encWmN = (float*)alloc(DCH * 16 * 4);
  float*  encbmN = (float*)alloc(DCH * 4);
  float*  encWmE = (float*)alloc(DCH * 8 * 4);
  float*  encbmE = (float*)alloc(DCH * 4);
  float*  foWm   = (float*)alloc(DCH * 4);
  float*  pb1m   = (float*)alloc((long long)L * 256 * 4);
  float*  pb2m   = (float*)alloc((long long)L * 128 * 4);
  __bf16* w1b    = (__bf16*)alloc((long long)L * 256 * 128 * 2);
  __bf16* w2b    = (__bf16*)alloc((long long)L * 128 * 256 * 2);
  __bf16* pw1b   = (__bf16*)alloc((long long)L * 256 * 128 * 2);
  __bf16* pw2b   = (__bf16*)alloc((long long)L * 128 * 256 * 2);
  float*  pooled = (float*)alloc((long long)G * DCH * 4);
  float*  cnt    = (float*)alloc((long long)G * 4);
  (void)ws_size; (void)n_in;

  // ---- prep: masked params + bf16 weights ----
  k_maskmul_f32<<<nblk(DCH * 16), 256, 0, stream>>>(enc_n_W, enc_n_mW, encWmN, DCH * 16);
  k_maskmul_f32<<<nblk(DCH), 256, 0, stream>>>(enc_n_b, enc_n_mb, encbmN, DCH);
  k_maskmul_f32<<<nblk(DCH * 8), 256, 0, stream>>>(enc_e_W, enc_e_mW, encWmE, DCH * 8);
  k_maskmul_f32<<<nblk(DCH), 256, 0, stream>>>(enc_e_b, enc_e_mb, encbmE, DCH);
  k_maskmul_f32<<<nblk(DCH), 256, 0, stream>>>(fo_W, fo_mW, foWm, DCH);
  k_maskmul_f32<<<nblk((long long)L * 256), 256, 0, stream>>>(mb1, mM1b, pb1m, (long long)L * 256);
  k_maskmul_f32<<<nblk((long long)L * 128), 256, 0, stream>>>(mb2, mM2b, pb2m, (long long)L * 128);
  const long long WSZ = (long long)L * 256 * 128;
  k_maskmul_bf16<<<nblk(WSZ), 256, 0, stream>>>(cW1, nullptr, w1b, WSZ);
  k_maskmul_bf16<<<nblk(WSZ), 256, 0, stream>>>(cW2, nullptr, w2b, WSZ);
  k_maskmul_bf16<<<nblk(WSZ), 256, 0, stream>>>(mW1, mM1w, pw1b, WSZ);
  k_maskmul_bf16<<<nblk(WSZ), 256, 0, stream>>>(mW2, mM2w, pw2b, WSZ);

  // ---- node encoder + BN -> h (f32) ----
  k_fill<<<1, 256, 0, stream>>>(stats, 0.f, 256);
  k_encode_stats<<<nblk(NH), 256, 0, stream>>>(x, encWmN, encbmN, NH, 16, stats);
  k_bn_finalize<<<1, DCH, 0, stream>>>(stats, bn_n_g, bn_n_b, 1.f / (float)N, ss, DCH);
  k_encode_apply<<<nblk(NH), 256, 0, stream>>>(x, encWmN, encbmN, ss, h, nullptr, NH, 16);

  // ---- edge encoder + BN -> e (bf16) ----
  k_fill<<<1, 256, 0, stream>>>(stats, 0.f, 256);
  k_encode_stats<<<nblk(EH), 256, 0, stream>>>(eattr, encWmE, encbmE, EH, 8, stats);
  k_bn_finalize<<<1, DCH, 0, stream>>>(stats, bn_e_g, bn_e_b, 1.f / (float)E, ss, DCH);
  k_encode_apply<<<nblk(EH), 256, 0, stream>>>(eattr, encWmE, encbmE, ss, nullptr, e_bf, EH, 8);

  // ---- layers ----
  const long long TE = (long long)E * 32;      // edge pass threads (4 ch/thread)
  const int eb = (int)((TE + 255) / 256);
  dim3 g1((N + 63) / 64, 256 / 32);            // GEMM to Nout=256
  dim3 g2((N + 63) / 64, 128 / 32);            // GEMM to Nout=128

  for (int l = 0; l < L; ++l) {
    // softmax aggregation
    k_init3<<<nblk(NH), 256, 0, stream>>>(mx, den, agg, NH);
    k_edge_max<<<eb, 256, 0, stream>>>(e_bf, h, src, dst, tvec, l, mx, TE);
    k_edge_den<<<eb, 256, 0, stream>>>(e_bf, h, src, dst, tvec, l, mx, den, TE);
    k_edge_agg<<<eb, 256, 0, stream>>>(e_bf, h, src, dst, tvec, l, mx, den, agg, TE);
    // u = agg + h  (bf16 for WMMA)
    k_add_bf16<<<nblk(NH), 256, 0, stream>>>(agg, h, u_bf, NH);

    // GENConv MLP: Linear(128->256) -> BN -> ReLU -> Linear(256->128)
    k_gemm_bf16<<<g1, 256, 0, stream>>>(u_bf, w1b + (long long)l * 256 * 128,
                                        cb1 + l * 256, t1, N, 128, 256);
    k_fill<<<1, 512, 0, stream>>>(stats, 0.f, 512);
    k_colstats<<<nblk((long long)N * 256), 256, 512 * 4, stream>>>(t1, (long long)N * 256, 256, stats);
    k_bn_finalize<<<1, 256, 0, stream>>>(stats, cg1 + l * 256, cbt1 + l * 256, 1.f / (float)N, ss, 256);
    k_bn_relu_bf16<<<nblk((long long)N * 256), 256, 0, stream>>>(t1, ss, t1_bf, (long long)N * 256, 256);
    k_gemm_bf16<<<g2, 256, 0, stream>>>(t1_bf, w2b + (long long)l * 128 * 256,
                                        cb2 + l * 128, t2, N, 256, 128);
    k_cvt_bf16<<<nblk(NH), 256, 0, stream>>>(t2, t2_bf, NH);

    // pruned MLP: (masked Linear -> BN -> ReLU) x 2, residual into h
    k_gemm_bf16<<<g1, 256, 0, stream>>>(t2_bf, pw1b + (long long)l * 256 * 128,
                                        pb1m + l * 256, t1, N, 128, 256);
    k_fill<<<1, 512, 0, stream>>>(stats, 0.f, 512);
    k_colstats<<<nblk((long long)N * 256), 256, 512 * 4, stream>>>(t1, (long long)N * 256, 256, stats);
    k_bn_finalize<<<1, 256, 0, stream>>>(stats, mg1 + l * 256, mbt1 + l * 256, 1.f / (float)N, ss, 256);
    k_bn_relu_bf16<<<nblk((long long)N * 256), 256, 0, stream>>>(t1, ss, t1_bf, (long long)N * 256, 256);
    k_gemm_bf16<<<g2, 256, 0, stream>>>(t1_bf, pw2b + (long long)l * 128 * 256,
                                        pb2m + l * 128, t4, N, 256, 128);
    k_fill<<<1, 512, 0, stream>>>(stats, 0.f, 512);
    k_colstats<<<nblk(NH), 256, 256 * 4, stream>>>(t4, NH, 128, stats);
    k_bn_finalize<<<1, DCH, 0, stream>>>(stats, mg2 + l * 128, mbt2 + l * 128, 1.f / (float)N, ss, DCH);
    k_bn_relu_add<<<nblk(NH), 256, 0, stream>>>(t4, ss, h, NH);
  }

  // ---- pooling + readout ----
  k_fill<<<nblk((long long)G * DCH), 256, 0, stream>>>(pooled, 0.f, (long long)G * DCH);
  k_fill<<<1, 256, 0, stream>>>(cnt, 0.f, G);
  k_pool<<<nblk(NH), 256, 0, stream>>>(h, batch, pooled, cnt, NH);
  k_final<<<(G + 7) / 8, 256, 0, stream>>>(pooled, cnt, foWm, fo_b, fo_mb, out, G);
}
